// LocalSlidingWindowAttention_62792421868023
// MI455X (gfx1250) — matrix operand order repaired and checked by reading, not compiled, not run on hardware
//
#include <hip/hip_runtime.h>
#include <hip/hip_bf16.h>

// ---------------------------------------------------------------------------
// LocalSlidingWindowAttention on gfx1250 (MI455X), wave32 + WMMA f16.
// B=2, T=2048, D=1024, H=16, DH=64, WINDOW=512.
// f16 working set (~48MB) is L2-resident (192MB) -> GEMMs load from global;
// A-fragment loads are B128, B frags per-element; weights pre-transposed to [KxN]; K written
// transposed [B,H,DH,T] by its projection GEMM so attention S-fragments are
// contiguous. LDS used only for the P-tile C->A relayout.
// ---------------------------------------------------------------------------

typedef __attribute__((ext_vector_type(16))) _Float16 v16h;
typedef __attribute__((ext_vector_type(8)))  _Float16 v8h;
typedef __attribute__((ext_vector_type(8)))  float    v8f;

#define BATCH   2
#define SEQ     2048
#define DMODEL  1024
#define NHEAD   16
#define HDIM    64
#define WIN     512
#define MROWS   (BATCH * SEQ)      // 4096
#define NCOLS   (NHEAD * HDIM)     // 1024

#define MASK_S  (-3.0e38f)
#define M_INIT  (-1.0e30f)

__device__ inline v16h join8(v8h a, v8h b) {
    return __builtin_shufflevector(a, b, 0, 1, 2, 3, 4, 5, 6, 7,
                                   8, 9, 10, 11, 12, 13, 14, 15);
}

// ---------------------------------------------------------------------------
// Fragment loaders (layouts per CDNA5 ISA 7.12.2, wave32). A frags B128.
// ---------------------------------------------------------------------------

// A 16x32 f16: lane row M = lane&15; per lane two contiguous 8-half runs:
// k in [hi*8, hi*8+8) and [16+hi*8, 16+hi*8+8).
__device__ inline v16h load_a_frag(const _Float16* __restrict__ p, int ld,
                                   int row0, int k0, int lane) {
    const int m  = row0 + (lane & 15);
    const int hi = lane >> 4;
    const _Float16* rp = p + (size_t)m * ld + k0 + hi * 8;
    const v8h lo = *(const v8h*)(rp);
    const v8h hh = *(const v8h*)(rp + 16);
    return join8(lo, hh);
}

// B 32x16 f16 from a row-major [K x N]-style source:
// lane holds column c0+(lane&15); k in [hi*8, hi*8+8) and [16+hi*8, 16+hi*8+8).
__device__ inline v16h load_b_frag(const _Float16* __restrict__ p, int ld,
                                   int r0, int c0, int lane) {
    const int n  = c0 + (lane & 15);
    const int hi = lane >> 4;
    v16h v;
#pragma unroll
    for (int s = 0; s < 16; ++s) {
        const int kk = r0 + ((s < 8) ? (hi * 8 + s) : (16 + hi * 8 + (s - 8)));
        v[s] = p[(size_t)kk * ld + n];
    }
    return v;
}

// Row-wise reductions across a 16-lane half (xor masks < 16 stay in-half).
__device__ inline float half_reduce_max(float v) {
#pragma unroll
    for (int m = 1; m < 16; m <<= 1) v = fmaxf(v, __shfl_xor(v, m, 32));
    return v;
}
__device__ inline float half_reduce_sum(float v) {
#pragma unroll
    for (int m = 1; m < 16; m <<= 1) v += __shfl_xor(v, m, 32);
    return v;
}

// ---------------------------------------------------------------------------
// fp32 -> fp16 conversion (plain) and transpose-convert for weights:
// Wt[k*N + n] = W[n*K + k]
// ---------------------------------------------------------------------------
__global__ __launch_bounds__(256) void cvt_f32_f16(const float* __restrict__ in,
                                                   _Float16* __restrict__ out, int n) {
    for (int i = blockIdx.x * 256 + threadIdx.x; i < n; i += gridDim.x * 256)
        out[i] = (_Float16)in[i];
}

__global__ __launch_bounds__(256) void cvt_transpose_f32_f16(
    const float* __restrict__ w, _Float16* __restrict__ wt, int N, int K) {
    for (int i = blockIdx.x * 256 + threadIdx.x; i < N * K; i += gridDim.x * 256) {
        const int k = i / N;          // output row
        const int n = i - k * N;      // output col (fastest -> coalesced store)
        wt[i] = (_Float16)w[(size_t)n * K + k];
    }
}

// ---------------------------------------------------------------------------
// GEMM: C[M,N] = A[M,K] * Bkn[K,N]   (2x2 WMMA tiles per wave, K-step 32)
// Optional outputs: f16 row-major, f16 K-transposed [B,H,DH,T], f32 row-major,
// and f32 sliding-window cache extraction.
// ---------------------------------------------------------------------------
__global__ __launch_bounds__(256) void gemm_wmma_f16(
    const _Float16* __restrict__ A, const _Float16* __restrict__ Bkn,
    _Float16* __restrict__ out_h, _Float16* __restrict__ out_kt,
    float* __restrict__ out_f, float* __restrict__ cache,
    int Mdim, int Ndim, int Kdim) {

    const int lane = threadIdx.x & 31;
    const int wid  = blockIdx.x * 8 + (threadIdx.x >> 5);
    const int ntn  = Ndim >> 5;                 // 32-wide macro tiles
    const int tm   = wid / ntn;
    const int tn   = wid - tm * ntn;
    if (tm * 32 >= Mdim) return;

    const int row0 = tm << 5;
    const int n0   = tn << 5;

    v8f acc00 = {}, acc01 = {}, acc10 = {}, acc11 = {};
    for (int k0 = 0; k0 < Kdim; k0 += 32) {
        __builtin_prefetch(A + (size_t)(row0 + (lane & 15)) * Kdim + k0 + 64, 0, 1);
        const v16h a0 = load_a_frag(A, Kdim, row0,      k0, lane);
        const v16h a1 = load_a_frag(A, Kdim, row0 + 16, k0, lane);
        const v16h b0 = load_b_frag(Bkn, Ndim, k0, n0,      lane);
        const v16h b1 = load_b_frag(Bkn, Ndim, k0, n0 + 16, lane);
        acc00 = __builtin_amdgcn_wmma_f32_16x16x32_f16(false, a0, false, b0,
                                                       (short)0, acc00, false, false);
        acc01 = __builtin_amdgcn_wmma_f32_16x16x32_f16(false, a0, false, b1,
                                                       (short)0, acc01, false, false);
        acc10 = __builtin_amdgcn_wmma_f32_16x16x32_f16(false, a1, false, b0,
                                                       (short)0, acc10, false, false);
        acc11 = __builtin_amdgcn_wmma_f32_16x16x32_f16(false, a1, false, b1,
                                                       (short)0, acc11, false, false);
    }

    const int hi = lane >> 4;
    const int cc = lane & 15;
#pragma unroll
    for (int sub = 0; sub < 4; ++sub) {
        const v8f* accp = (sub == 0) ? &acc00 : (sub == 1) ? &acc01
                         : (sub == 2) ? &acc10 : &acc11;
        const int r0 = row0 + ((sub >> 1) << 4);
        const int c0 = n0 + ((sub & 1) << 4);
        const int col = c0 + cc;
#pragma unroll
        for (int r = 0; r < 8; ++r) {
            const int row = r0 + r + hi * 8;
            const float v = (*accp)[r];
            if (out_h)  out_h[(size_t)row * Ndim + col] = (_Float16)v;
            if (out_f)  out_f[(size_t)row * Ndim + col] = v;
            if (out_kt) {
                const int b = row >> 11;            // / SEQ
                const int t = row & (SEQ - 1);
                const int h = col >> 6;             // / HDIM
                const int d = col & (HDIM - 1);
                out_kt[((size_t)(b * NHEAD + h) * HDIM + d) * SEQ + t] = (_Float16)v;
            }
            if (cache) {
                const int b = row >> 11;
                const int t = row & (SEQ - 1);
                if (t >= SEQ - WIN) {
                    const int w = t - (SEQ - WIN);
                    const int h = col >> 6;
                    const int d = col & (HDIM - 1);
                    cache[(((size_t)(b * NHEAD + h) * WIN) + w) * HDIM + d] = v;
                }
            }
        }
    }
}

// ---------------------------------------------------------------------------
// Flash attention with causal sliding window (512), one 16-query tile / wave.
// S = Q Kt via WMMA (Kt pre-transposed -> lane-coalesced B loads),
// online softmax (xor-shuffle row reductions), P re-laid out via LDS,
// O += P V via WMMA (accumulator as C operand).
// ---------------------------------------------------------------------------
__global__ __launch_bounds__(256) void attn_wmma(
    const _Float16* __restrict__ Qh, const _Float16* __restrict__ Kt,
    const _Float16* __restrict__ Vh, _Float16* __restrict__ Oh) {

    __shared__ __align__(16) _Float16 plds[8][16 * 32]; // per-wave P tile

    const int lane = threadIdx.x & 31;
    const int widx = threadIdx.x >> 5;
    const int wid  = blockIdx.x * 8 + widx;
    const int qt_per = SEQ >> 4;                  // 128
    const int bh   = wid / qt_per;
    const int qt   = wid - bh * qt_per;
    const int b    = bh >> 4;                     // / NHEAD
    const int q0   = qt << 4;
    const int rbase = b * SEQ;                    // row offset of this batch
    const int c0    = (bh & (NHEAD - 1)) * HDIM;  // column offset of this head
    const int ktr0  = bh * HDIM;                  // Kt row block for this (b,h)

    // Q fragments for the two K=32 chunks of DH=64 (reused across key blocks)
    const v16h aq0 = load_a_frag(Qh, NCOLS, rbase + q0, c0,      lane);
    const v16h aq1 = load_a_frag(Qh, NCOLS, rbase + q0, c0 + 32, lane);

    v8f o0 = {}, o1 = {}, o2 = {}, o3 = {};
    float mrow[8], lrow[8];
#pragma unroll
    for (int r = 0; r < 8; ++r) { mrow[r] = M_INIT; lrow[r] = 0.0f; }

    const int hi = lane >> 4;
    int lo = q0 - (WIN - 1);
    if (lo < 0) lo = 0;
    lo &= ~31;

    for (int kb = lo; kb <= q0 + 15; kb += 32) {
        __builtin_prefetch(Vh + (size_t)(rbase + kb + 32 + (lane & 15)) * NCOLS + c0, 0, 1);

        // ---- S tiles: keys [kb, kb+16) and [kb+16, kb+32) -----------------
        // B[kk][n] = Kt[ktr0 + chunk + kk][kb + n] : contiguous in n.
        v8f s0 = {}, s1 = {};
        {
            v16h bk = load_b_frag(Kt, SEQ, ktr0,      kb,      lane);
            s0 = __builtin_amdgcn_wmma_f32_16x16x32_f16(false, aq0, false, bk,
                                                        (short)0, s0, false, false);
            bk = load_b_frag(Kt, SEQ, ktr0 + 32, kb,      lane);
            s0 = __builtin_amdgcn_wmma_f32_16x16x32_f16(false, aq1, false, bk,
                                                        (short)0, s0, false, false);
            bk = load_b_frag(Kt, SEQ, ktr0,      kb + 16, lane);
            s1 = __builtin_amdgcn_wmma_f32_16x16x32_f16(false, aq0, false, bk,
                                                        (short)0, s1, false, false);
            bk = load_b_frag(Kt, SEQ, ktr0 + 32, kb + 16, lane);
            s1 = __builtin_amdgcn_wmma_f32_16x16x32_f16(false, aq1, false, bk,
                                                        (short)0, s1, false, false);
        }

        // ---- scale + sliding-window causal mask ---------------------------
        const int kc0 = kb + (lane & 15);
        const int kc1 = kc0 + 16;
#pragma unroll
        for (int r = 0; r < 8; ++r) {
            const int q = q0 + r + hi * 8;
            const bool v0 = (kc0 <= q) && (q - kc0 < WIN);
            const bool v1 = (kc1 <= q) && (q - kc1 < WIN);
            s0[r] = v0 ? s0[r] * 0.125f : MASK_S;   // 1/sqrt(64)
            s1[r] = v1 ? s1[r] * 0.125f : MASK_S;
        }

        // ---- online softmax update ----------------------------------------
        float fscale[8];
#pragma unroll
        for (int r = 0; r < 8; ++r) {
            const float rmax = half_reduce_max(fmaxf(s0[r], s1[r]));
            const float mnew = fmaxf(mrow[r], rmax);
            fscale[r] = __expf(mrow[r] - mnew);
            mrow[r]   = mnew;
            const float p0 = __expf(s0[r] - mnew);
            const float p1 = __expf(s1[r] - mnew);
            s0[r] = p0; s1[r] = p1;
            lrow[r] = lrow[r] * fscale[r] + half_reduce_sum(p0 + p1);
        }
#pragma unroll
        for (int r = 0; r < 8; ++r) {
            o0[r] *= fscale[r]; o1[r] *= fscale[r];
            o2[r] *= fscale[r]; o3[r] *= fscale[r];
        }

        // ---- P: C-layout -> LDS (row-major 16x32) -> A-layout -------------
        {
            _Float16* pt = plds[widx];
            const int cc = lane & 15;
#pragma unroll
            for (int r = 0; r < 8; ++r) {
                const int row = r + hi * 8;
                pt[row * 32 + cc]      = (_Float16)s0[r];
                pt[row * 32 + cc + 16] = (_Float16)s1[r];
            }
        }
        const v16h pa = load_a_frag(plds[widx], 32, 0, 0, lane); // 2x ds_load_b128

        // ---- O += P * V  (4 N-tiles of 16 over DH=64) ---------------------
        v16h bv = load_b_frag(Vh, NCOLS, rbase + kb, c0, lane);
        o0 = __builtin_amdgcn_wmma_f32_16x16x32_f16(false, pa, false, bv,
                                                    (short)0, o0, false, false);
        bv = load_b_frag(Vh, NCOLS, rbase + kb, c0 + 16, lane);
        o1 = __builtin_amdgcn_wmma_f32_16x16x32_f16(false, pa, false, bv,
                                                    (short)0, o1, false, false);
        bv = load_b_frag(Vh, NCOLS, rbase + kb, c0 + 32, lane);
        o2 = __builtin_amdgcn_wmma_f32_16x16x32_f16(false, pa, false, bv,
                                                    (short)0, o2, false, false);
        bv = load_b_frag(Vh, NCOLS, rbase + kb, c0 + 48, lane);
        o3 = __builtin_amdgcn_wmma_f32_16x16x32_f16(false, pa, false, bv,
                                                    (short)0, o3, false, false);
    }

    // ---- normalize and write O (f16) --------------------------------------
    const int cc = lane & 15;
#pragma unroll
    for (int r = 0; r < 8; ++r) {
        const float inv = 1.0f / lrow[r];
        const size_t row = (size_t)(rbase + q0 + r + hi * 8) * NCOLS + c0;
        Oh[row + cc]      = (_Float16)(o0[r] * inv);
        Oh[row + cc + 16] = (_Float16)(o1[r] * inv);
        Oh[row + cc + 32] = (_Float16)(o2[r] * inv);
        Oh[row + cc + 48] = (_Float16)(o3[r] * inv);
    }
}

// ---------------------------------------------------------------------------
// Host-side orchestration
// ---------------------------------------------------------------------------
extern "C" void kernel_launch(void* const* d_in, const int* in_sizes, int n_in,
                              void* d_out, int out_size, void* d_ws, size_t ws_size,
                              hipStream_t stream) {
    (void)in_sizes; (void)n_in; (void)out_size; (void)ws_size;

    const float* x  = (const float*)d_in[0];
    const float* Wq = (const float*)d_in[1];
    const float* Wk = (const float*)d_in[2];
    const float* Wv = (const float*)d_in[3];
    const float* Wo = (const float*)d_in[4];

    float* y  = (float*)d_out;                                   // [B,T,D]
    float* kc = y + (size_t)BATCH * SEQ * DMODEL;                // [B,H,W,DH]
    float* vc = kc + (size_t)BATCH * NHEAD * WIN * HDIM;

    char* ws = (char*)d_ws;
    const size_t MB = 1024 * 1024;
    _Float16* xh  = (_Float16*)(ws + 0 * MB);   // 8MB  [M x K]
    _Float16* wqt = (_Float16*)(ws + 8 * MB);   // 2MB each, [K x N]
    _Float16* wkt = (_Float16*)(ws + 10 * MB);
    _Float16* wvt = (_Float16*)(ws + 12 * MB);
    _Float16* wot = (_Float16*)(ws + 14 * MB);
    _Float16* Qh  = (_Float16*)(ws + 16 * MB);  // 8MB [M x N] row-major
    _Float16* Kt  = (_Float16*)(ws + 24 * MB);  // 8MB [B,H,DH,T] transposed
    _Float16* Vh  = (_Float16*)(ws + 32 * MB);  // 8MB [M x N] row-major
    _Float16* Oh  = (_Float16*)(ws + 40 * MB);  // 8MB [M x N] row-major

    // 1) fp32 -> fp16 (weights transposed to [K x N] for contiguous B frags)
    cvt_f32_f16<<<1024, 256, 0, stream>>>(x, xh, MROWS * DMODEL);
    cvt_transpose_f32_f16<<<512, 256, 0, stream>>>(Wq, wqt, NCOLS, DMODEL);
    cvt_transpose_f32_f16<<<512, 256, 0, stream>>>(Wk, wkt, NCOLS, DMODEL);
    cvt_transpose_f32_f16<<<512, 256, 0, stream>>>(Wv, wvt, NCOLS, DMODEL);
    cvt_transpose_f32_f16<<<512, 256, 0, stream>>>(Wo, wot, DMODEL, NCOLS);

    // 2) Q/K/V projections. K writes its transposed layout + f32 cache,
    //    V writes row-major + f32 cache, straight from f32 accumulators.
    const int gemm_blocks = (MROWS / 32) * (NCOLS / 32) / 8;     // 512
    gemm_wmma_f16<<<gemm_blocks, 256, 0, stream>>>(
        xh, wqt, Qh, nullptr, nullptr, nullptr, MROWS, NCOLS, DMODEL);
    gemm_wmma_f16<<<gemm_blocks, 256, 0, stream>>>(
        xh, wkt, nullptr, Kt, nullptr, kc, MROWS, NCOLS, DMODEL);
    gemm_wmma_f16<<<gemm_blocks, 256, 0, stream>>>(
        xh, wvt, Vh, nullptr, nullptr, vc, MROWS, NCOLS, DMODEL);

    // 3) sliding-window flash attention
    const int attn_blocks = BATCH * NHEAD * (SEQ / 16) / 8;      // 512
    attn_wmma<<<attn_blocks, 256, 0, stream>>>(Qh, Kt, Vh, Oh);

    // 4) output projection -> fp32 y
    gemm_wmma_f16<<<gemm_blocks, 256, 0, stream>>>(
        Oh, wot, nullptr, nullptr, y, nullptr, MROWS, NCOLS, DMODEL);
}
